// DPGN_75436805587537
// MI455X (gfx1250) — matrix operand order repaired and checked by reading, not compiled, not run on hardware
//
#include <hip/hip_runtime.h>
#include <hip/hip_bf16.h>

#define DEVI __device__ __forceinline__

namespace {
constexpr int B_ = 16, N_ = 160, C_ = 64, C2_ = 128, S_ = 80;
constexpr int TM = 32;            // rows (i,j)-pairs per workgroup
constexpr int APAD = 16;          // bf16 padding to keep 16B alignment + spread banks
constexpr int SLOTS = 64;         // replication factor for stat atomics
constexpr int NN = N_ * N_;
constexpr int MTOT = B_ * NN;     // 409600 GEMM rows
constexpr int NWG = MTOT / TM;    // 12800 workgroups
constexpr float SLOPE = 0.01f;
constexpr float BN_EPS = 1e-5f;
}

typedef __attribute__((ext_vector_type(16))) __bf16 v16bf;
typedef __attribute__((ext_vector_type(8)))  __bf16 v8bf;
typedef __attribute__((ext_vector_type(8)))  float  v8f;

DEVI __bf16 f2bf(float f) {
  unsigned int u = __builtin_bit_cast(unsigned int, f);
  u += 0x7fffu + ((u >> 16) & 1u);           // round-to-nearest-even
  unsigned short h = (unsigned short)(u >> 16);
  return __builtin_bit_cast(__bf16, h);
}

DEVI float lrelu(float x) { return x >= 0.f ? x : SLOPE * x; }

// Async global->LDS copy of one 16B chunk per lane (GLOBAL_LOAD_ASYNC_TO_LDS_B128,
// tracked by ASYNCcnt). vdst VGPR carries the LDS byte address, vaddr the 64-bit
// global address (GV mode, saddr=off).
DEVI void async_copy_b128(unsigned int lds_addr, unsigned long long gaddr) {
  asm volatile("global_load_async_to_lds_b128 %0, %1, off"
               :: "v"(lds_addr), "v"(gaddr)
               : "memory");
}

DEVI void wait_asynccnt0() {
  asm volatile("s_wait_asynccnt 0x0" ::: "memory");
}

// A-matrix 16x32 bf16 fragment (ISA 7.12.2): lanes<16 row=lane, K {kk..kk+7, kk+16..kk+23};
// lanes>=16 row=lane-16, K {kk+8..kk+15, kk+24..kk+31}. Two 16B vector loads.
DEVI v16bf load_a_frag(const __bf16* base, int ld, int m0, int kk, int lane) {
  const int r = lane & 15, half = lane >> 4;
  const __bf16* p = base + (m0 + r) * ld + kk + half * 8;
  union { v16bf v; v8bf h[2]; } u;
  u.h[0] = *(const v8bf*)(p);
  u.h[1] = *(const v8bf*)(p + 16);
  return u.v;
}

// B-matrix 32x16 bf16 fragment: lane<16 col=lane holds K {kk..kk+7, kk+16..kk+23}; lane>=16
// col=lane-16 holds K {kk+8..kk+15, kk+24..kk+31} (W stored [O][K] row-major).
DEVI v16bf load_b_frag(const __bf16* W, int ld, int o0, int kk, int lane) {
  const int col = lane & 15, half = lane >> 4;
  const __bf16* p = W + (o0 + col) * ld + kk + half * 8;
  union { v16bf v; v8bf h[2]; } u;
  u.h[0] = *(const v8bf*)(p);
  u.h[1] = *(const v8bf*)(p + 16);
  return u.v;
}

__global__ void prep_w(const float* __restrict__ w1, const float* __restrict__ w2,
                       __bf16* __restrict__ w1bf, __bf16* __restrict__ w2bf) {
  int i = blockIdx.x * 256 + threadIdx.x;
  if (i < C2_ * C_) { w1bf[i] = f2bf(w1[i]); w2bf[i] = f2bf(w2[i]); }
}

__global__ void zero_f32(float* __restrict__ p, int n) {
  int i = blockIdx.x * 256 + threadIdx.x;
  if (i < n) p[i] = 0.f;
}

// Fused point_similarity worker. One 256-thread (8-wave) WG handles 32 rows = (b,i,j0..j0+31).
// PASS 1: GEMM1 (sim@W1), accumulate per-channel sum/sumsq (BN1 stats).
// PASS 2: GEMM1 + BN1 + LReLU -> LDS -> GEMM2 (@W2), accumulate BN2 stats.
// PASS 3: full chain + BN2 + LReLU + dot(w3) + sigmoid -> e[b,i,j].
template <int PASS>
__global__ __launch_bounds__(256) void ps_fused(
    const float* __restrict__ V,        // [B,N,C]
    const __bf16* __restrict__ W1bf,    // [2C,C]
    const __bf16* __restrict__ W2bf,    // [C,2C]
    const float* __restrict__ bn1,      // scale[2C], shift[2C]
    const float* __restrict__ bn2,      // scale[C], shift[C]
    const float* __restrict__ w3,       // [C]
    const float* __restrict__ b3p,      // [1]
    float* __restrict__ stats1,         // [SLOTS][2][2C]
    float* __restrict__ stats2,         // [SLOTS][2][C]
    float* __restrict__ ebuf)           // [B,N,N]
{
  __shared__ __attribute__((aligned(16))) __bf16 sA[TM][C_ + APAD];    // sim tile (bf16)
  __shared__ __attribute__((aligned(16))) __bf16 sH[TM][C2_ + APAD];   // act1 tile (bf16)
  __shared__ __attribute__((aligned(16))) __bf16 sW1[C2_ * C_];        // W1 staged (16KB)
  __shared__ __attribute__((aligned(16))) __bf16 sW2[C_ * C2_];        // W2 staged (16KB)
  __shared__ float eacc[TM];

  const int wg = blockIdx.x;
  const int m0g = wg * TM;
  const int b = m0g / NN;
  const int rem = m0g - b * NN;
  const int i = rem / N_;
  const int j0 = rem - i * N_;           // 32 | N, so one (b,i) per WG
  const int t = threadIdx.x;
  const int lane = t & 31;
  const int wave = t >> 5;
  const int slot = wg & (SLOTS - 1);

  // ---- kick off async weight staging into LDS; overlaps with sim-tile build
  {
    constexpr int NCHUNK = (C2_ * C_ * 2) / 16;        // 1024 x 16B chunks
    const unsigned int lds1 = (unsigned int)(size_t)(&sW1[0]);
    const unsigned long long g1a = (unsigned long long)(size_t)W1bf;
    for (int cidx = t; cidx < NCHUNK; cidx += 256)     // NCHUNK % 256 == 0: uniform
      async_copy_b128(lds1 + cidx * 16, g1a + (unsigned long long)cidx * 16);
    if (PASS >= 2) {
      const unsigned int lds2 = (unsigned int)(size_t)(&sW2[0]);
      const unsigned long long g2a = (unsigned long long)(size_t)W2bf;
      for (int cidx = t; cidx < NCHUNK; cidx += 256)
        async_copy_b128(lds2 + cidx * 16, g2a + (unsigned long long)cidx * 16);
    }
  }

  // ---- build sim tile: sim[r][c] = (v[b,i,c] - v[b,j0+r,c])^2, bf16 in LDS
  const float* vi = V + (size_t)(b * N_ + i) * C_;
  const float* vj = V + (size_t)(b * N_ + j0) * C_;
  for (int idx = t; idx < TM * C_; idx += 256) {
    int r = idx >> 6, c = idx & (C_ - 1);
    float d = vi[c] - vj[r * C_ + c];
    sA[r][c] = f2bf(d * d);
  }
  if (PASS == 3 && t < TM) eacc[t] = 0.f;
  wait_asynccnt0();                      // wave's async LDS writes complete
  __syncthreads();                       // publish to all waves

  // ---- GEMM1: [32 x 64] x [64 x 128] -> 16 wave-tiles over 8 waves (2 reps)
  for (int rep = 0; rep < 2; ++rep) {
    const int tile = wave + 8 * rep;
    const int mt = tile & 1;             // 16-row half
    const int o0 = (tile >> 1) * 16;     // output-channel tile
    v8f acc = {};
#pragma unroll
    for (int kk = 0; kk < C_; kk += 32) {
      v16bf a  = load_a_frag(&sA[0][0], C_ + APAD, mt * 16, kk, lane);
      v16bf bb = load_b_frag(&sW1[0], C_, o0, kk, lane);
      acc = __builtin_amdgcn_wmma_f32_16x16x32_bf16(false, a, false, bb,
                                                    (short)0, acc, false, false);
    }
    const int ch = o0 + (lane & 15);
    const int half = lane >> 4;
    if (PASS == 1) {
      float s = 0.f, q = 0.f;
#pragma unroll
      for (int v = 0; v < 8; ++v) { float x = acc[v]; s += x; q += x * x; }
      s += __shfl_xor(s, 16);
      q += __shfl_xor(q, 16);
      if (half == 0) {
        atomicAdd(&stats1[(slot * 2 + 0) * C2_ + ch], s);
        atomicAdd(&stats1[(slot * 2 + 1) * C2_ + ch], q);
      }
    } else {
      const float sc = bn1[ch], sh = bn1[C2_ + ch];
#pragma unroll
      for (int v = 0; v < 8; ++v) {
        float x = lrelu(acc[v] * sc + sh);
        sH[mt * 16 + v + 8 * half][ch] = f2bf(x);   // C-layout -> row-major LDS
      }
    }
  }
  if (PASS == 1) return;
  __syncthreads();

  // ---- GEMM2: [32 x 128] x [128 x 64] -> 8 wave-tiles, one per wave
  const int mt = wave & 1;
  const int o0 = (wave >> 1) * 16;
  v8f acc = {};
#pragma unroll
  for (int kk = 0; kk < C2_; kk += 32) {
    v16bf a  = load_a_frag(&sH[0][0], C2_ + APAD, mt * 16, kk, lane);
    v16bf bb = load_b_frag(&sW2[0], C2_, o0, kk, lane);
    acc = __builtin_amdgcn_wmma_f32_16x16x32_bf16(false, a, false, bb,
                                                  (short)0, acc, false, false);
  }
  const int ch = o0 + (lane & 15);
  const int half = lane >> 4;
  if (PASS == 2) {
    float s = 0.f, q = 0.f;
#pragma unroll
    for (int v = 0; v < 8; ++v) { float x = acc[v]; s += x; q += x * x; }
    s += __shfl_xor(s, 16);
    q += __shfl_xor(q, 16);
    if (half == 0) {
      atomicAdd(&stats2[(slot * 2 + 0) * C_ + ch], s);
      atomicAdd(&stats2[(slot * 2 + 1) * C_ + ch], q);
    }
    return;
  }

  // ---- PASS 3: BN2 + LReLU + dot(w3) via LDS f32 atomics, then sigmoid
  const float sc = bn2[ch], sh = bn2[C_ + ch];
  const float wv = w3[ch];
#pragma unroll
  for (int v = 0; v < 8; ++v) {
    float x = lrelu(acc[v] * sc + sh);
    atomicAdd(&eacc[mt * 16 + v + 8 * half], x * wv);
  }
  __syncthreads();
  if (t < TM) {
    float z = eacc[t] + b3p[0];
    ebuf[(size_t)(b * N_ + i) * N_ + j0 + t] = 1.f / (1.f + __expf(-z));
  }
}

__global__ void finalize_bn(const float* __restrict__ stats, int nch,
                            const float* __restrict__ g, const float* __restrict__ bb,
                            float* __restrict__ bnout, float inv_count) {
  int ch = threadIdx.x;
  if (ch >= nch) return;
  float s = 0.f, q = 0.f;
  for (int sl = 0; sl < SLOTS; ++sl) {
    s += stats[(sl * 2 + 0) * nch + ch];
    q += stats[(sl * 2 + 1) * nch + ch];
  }
  float m = s * inv_count;
  float var = fmaxf(q * inv_count - m * m, 0.f);
  float inv = rsqrtf(var + BN_EPS);
  float scale = inv * g[ch];
  bnout[ch] = scale;
  bnout[nch + ch] = bb[ch] - m * scale;
}

// Per-(b,i) edge renormalization (reference tail of point_similarity).
__global__ __launch_bounds__(256) void edge_norm(const float* __restrict__ e,
                                                 const float* __restrict__ ep_last,
                                                 float* __restrict__ out) {
  __shared__ float red[256];
  __shared__ float sc[3];
  const int bi = blockIdx.x;            // b*N + i
  const int i = bi % N_;
  const int t = threadIdx.x;
  const float* epr = ep_last + (size_t)bi * N_;
  const float* er  = e + (size_t)bi * N_;

  float epm = 0.f, x = 0.f;
  if (t < N_) {
    float ep = epr[t];
    epm = (t == i) ? 0.f : ep;
    x = er[t] * epm;
  }
  red[t] = epm; __syncthreads();
  for (int s = 128; s > 0; s >>= 1) { if (t < s) red[t] += red[t + s]; __syncthreads(); }
  if (t == 0) sc[0] = red[0];
  __syncthreads();
  red[t] = fabsf(x); __syncthreads();
  for (int s = 128; s > 0; s >>= 1) { if (t < s) red[t] += red[t + s]; __syncthreads(); }
  if (t == 0) sc[1] = red[0];
  __syncthreads();
  const float rowsum = sc[0];
  const float l1 = fmaxf(sc[1], 1e-12f);
  float y = 0.f;
  if (t < N_) y = x / l1 * rowsum + ((t == i) ? 1.f : 0.f) + 1e-6f;
  red[t] = (t < N_) ? y : 0.f; __syncthreads();
  for (int s = 128; s > 0; s >>= 1) { if (t < s) red[t] += red[t + s]; __syncthreads(); }
  if (t == 0) sc[2] = red[0];
  __syncthreads();
  if (t < N_) out[(size_t)bi * N_ + t] = y / sc[2];
}

// P2DAgg: dn_out[b,n,o] = lrelu( sum_s concat(pe[b,n,:S], dn_in[b,n,:]) * w[o,:] + bias[o] )
__global__ void p2d_agg(const float* __restrict__ pe, const float* __restrict__ dn_in,
                        const float* __restrict__ w, const float* __restrict__ bias,
                        float* __restrict__ dn_out) {
  int idx = blockIdx.x * 256 + threadIdx.x;
  if (idx >= B_ * N_ * S_) return;
  const int o = idx % S_;
  const int bn = idx / S_;
  const float* per = pe + (size_t)bn * N_;     // first S entries of edge row
  const float* dnr = dn_in + (size_t)bn * S_;
  const float* wr = w + (size_t)o * (2 * S_);
  float acc = bias[o];
  for (int s = 0; s < S_; ++s) acc += per[s] * wr[s];
  for (int s = 0; s < S_; ++s) acc += dnr[s] * wr[S_ + s];
  dn_out[idx] = lrelu(acc);
}

extern "C" void kernel_launch(void* const* d_in, const int* in_sizes, int n_in,
                              void* d_out, int out_size, void* d_ws, size_t ws_size,
                              hipStream_t stream) {
  const float* middle = (const float*)d_in[0];
  const float* point  = (const float*)d_in[1];
  const float* dnode  = (const float*)d_in[2];
  // d_in[3] distribution_edge: unused by the reference forward
  const float* pedge  = (const float*)d_in[4];
  const float* w1 = (const float*)d_in[5];
  const float* g1 = (const float*)d_in[6];
  const float* b1 = (const float*)d_in[7];
  const float* w2 = (const float*)d_in[8];
  const float* g2 = (const float*)d_in[9];
  const float* b2 = (const float*)d_in[10];
  const float* w3 = (const float*)d_in[11];
  const float* b3 = (const float*)d_in[12];
  const float* p2dw = (const float*)d_in[13];
  const float* p2db = (const float*)d_in[14];
  float* out = (float*)d_out;

  // workspace carve-up (~5 MB of f32)
  float* peA    = (float*)d_ws;
  float* peB    = peA + MTOT;
  float* ebuf   = peB + MTOT;
  float* stats1 = ebuf + MTOT;                 // SLOTS*2*C2
  float* stats2 = stats1 + SLOTS * 2 * C2_;    // SLOTS*2*C
  float* bn1    = stats2 + SLOTS * 2 * C_;     // 2*C2
  float* bn2    = bn1 + 2 * C2_;               // 2*C
  __bf16* w1bf  = (__bf16*)(bn2 + 2 * C_);     // C2*C bf16
  __bf16* w2bf  = w1bf + C2_ * C_;             // C*C2 bf16

  prep_w<<<(C2_ * C_ + 255) / 256, 256, 0, stream>>>(w1, w2, w1bf, w2bf);

  const float invM = 1.0f / (float)MTOT;
  auto run_ps = [&](const float* v, const float* ep, float* pe_out) {
    zero_f32<<<(SLOTS * 2 * C2_ + 255) / 256, 256, 0, stream>>>(stats1, SLOTS * 2 * C2_);
    ps_fused<1><<<NWG, 256, 0, stream>>>(v, w1bf, w2bf, bn1, bn2, w3, b3, stats1, stats2, ebuf);
    finalize_bn<<<1, C2_, 0, stream>>>(stats1, C2_, g1, b1, bn1, invM);
    zero_f32<<<(SLOTS * 2 * C_ + 255) / 256, 256, 0, stream>>>(stats2, SLOTS * 2 * C_);
    ps_fused<2><<<NWG, 256, 0, stream>>>(v, w1bf, w2bf, bn1, bn2, w3, b3, stats1, stats2, ebuf);
    finalize_bn<<<1, C_, 0, stream>>>(stats2, C_, g2, b2, bn2, invM);
    ps_fused<3><<<NWG, 256, 0, stream>>>(v, w1bf, w2bf, bn1, bn2, w3, b3, stats1, stats2, ebuf);
    edge_norm<<<B_ * N_, 256, 0, stream>>>(ebuf, ep, pe_out);
  };

  // pe = ps(middle_node, point_edge)
  run_ps(middle, pedge, peA);
  // gen 0: pe = ps(point_node, pe); dn0 = p2d(pe, distribution_node)
  run_ps(point, peA, peB);
  p2d_agg<<<(B_ * N_ * S_ + 255) / 256, 256, 0, stream>>>(peB, dnode, p2dw, p2db, out);
  // gen 1: pe = ps(point_node, pe); dn1 = p2d(pe, dn0)
  run_ps(point, peB, peA);
  p2d_agg<<<(B_ * N_ * S_ + 255) / 256, 256, 0, stream>>>(
      peA, out, p2dw + (size_t)S_ * 2 * S_, p2db + S_, out + (size_t)B_ * N_ * S_);
}
